// PureSNN_16234976379199
// MI455X (gfx1250) — matrix operand order repaired and checked by reading, not compiled, not run on hardware
//
#include <hip/hip_runtime.h>
#include <stdint.h>

typedef _Float16 half_t;
typedef __attribute__((ext_vector_type(16))) half_t v16h;
typedef __attribute__((ext_vector_type(8)))  half_t v8h;
typedef __attribute__((ext_vector_type(4)))  half_t v4h;
typedef __attribute__((ext_vector_type(8)))  float  v8f;
typedef __attribute__((ext_vector_type(4)))  unsigned int u32x4;
typedef __attribute__((ext_vector_type(4)))  int i32x4;
typedef __attribute__((ext_vector_type(8)))  int i32x8;

constexpr int IN_DIM      = 784;
constexpr int H_DIM       = 128;
constexpr int OUT_DIM     = 10;
constexpr int T_STEPS     = 20;
constexpr int ROWS_PER_WG = 128;   // 8 waves x 16 rows

// ---------- fragment builders ----------
// A-matrix 16x32 f16 (ISA 7.12.2): lanes 0-15 hold K kbase..+7 / kbase+16..+23 (kbase=k0),
// lanes 16-31 the same with kbase=k0+8.
__device__ __forceinline__ v16h a_frag_global(const float* __restrict__ row, int kbase, bool full) {
  v16h r;
  const float4* p = reinterpret_cast<const float4*>(row + kbase);
  float4 x0 = p[0], x1 = p[1];
  r[0] = (half_t)x0.x; r[1] = (half_t)x0.y; r[2] = (half_t)x0.z; r[3] = (half_t)x0.w;
  r[4] = (half_t)x1.x; r[5] = (half_t)x1.y; r[6] = (half_t)x1.z; r[7] = (half_t)x1.w;
  if (full) {
    const float4* q = reinterpret_cast<const float4*>(row + kbase + 16);
    float4 y0 = q[0], y1 = q[1];
    r[8]  = (half_t)y0.x; r[9]  = (half_t)y0.y; r[10] = (half_t)y0.z; r[11] = (half_t)y0.w;
    r[12] = (half_t)y1.x; r[13] = (half_t)y1.y; r[14] = (half_t)y1.z; r[15] = (half_t)y1.w;
  } else {
#pragma unroll
    for (int j = 8; j < 16; ++j) r[j] = (half_t)0.0f;
  }
  return r;
}

// B 32x16 f16 in this code's contiguous k order (the s1 A fragments match): lane holds N=lane&15; lanes 0-15 K k0..k0+15, lanes 16-31 K k0+16..k0+31
// -> 16 contiguous f16 per lane.
__device__ __forceinline__ v16h b_frag_lds16(const half_t* p) {
  v8h lo = *reinterpret_cast<const v8h*>(p);
  v8h hi = *reinterpret_cast<const v8h*>(p + 8);
  return __builtin_shufflevector(lo, hi, 0,1,2,3,4,5,6,7,8,9,10,11,12,13,14,15);
}

__device__ __forceinline__ v16h a_frag_lds(const half_t* p, int kbase) {
  v8h lo = *reinterpret_cast<const v8h*>(p + kbase);
  v8h hi = *reinterpret_cast<const v8h*>(p + kbase + 16);
  return __builtin_shufflevector(lo, hi, 0,1,2,3,4,5,6,7,8,9,10,11,12,13,14,15);
}

__global__ __launch_bounds__(256)
void snn_fused_kernel(const float* __restrict__ x,  const float* __restrict__ W1,
                      const float* __restrict__ b1, const float* __restrict__ W2,
                      const float* __restrict__ b2, const float* __restrict__ Wr,
                      const float* __restrict__ br, float* __restrict__ out) {
  // TDM target for fp32 W2 (phase 0/1), reused as per-wave spike staging (phase 2/3)
  __shared__ __align__(16) union SharedA {
    float  w2stage[H_DIM * H_DIM];     // 64 KB
    half_t s1[8 * 16 * H_DIM];         // 32 KB
  } shA;
  __shared__ __align__(16) half_t w2lds[H_DIM * H_DIM];    // 32 KB, W2 as f16 [n][k]
  __shared__ __align__(16) half_t w1buf[2][H_DIM * 32];    // 16 KB, W1 chunk double buffer

  const int tid   = threadIdx.x;
  const int wave  = tid >> 5;
  const int lane  = tid & 31;
  const int lhalf = lane >> 4;
  const int lidx  = lane & 15;

  // ---- kick off async TDM load of W2 (128x128 f32, 64KB) into LDS; overlaps phase 1 ----
  if (wave == 0) {
    const uint64_t ga = (uint64_t)(uintptr_t)W2;
    const uint32_t ldsoff = (uint32_t)(uintptr_t)(&shA.w2stage[0]); // flat[31:0] == LDS offset
    u32x4 g0;
    g0[0] = 1u;                                                  // count=1, user descriptor
    g0[1] = ldsoff;                                              // lds_addr
    g0[2] = (uint32_t)ga;                                        // global_addr[31:0]
    g0[3] = ((uint32_t)(ga >> 32) & 0x01FFFFFFu) | 0x80000000u;  // global_addr[56:32] | type=2
    i32x8 g1;
    g1[0] = 0x00020000;          // workgroup_mask=0, data_size=2 (4 bytes)
    g1[1] = (int)(128u << 16);   // tensor_dim0 = 128 (bits 79:48)
    g1[2] = (int)(128u << 16);   // tensor_dim1 = 128 (bits 111:80)
    g1[3] = (int)(128u << 16);   // tile_dim0   = 128 (bits 127:112)
    g1[4] = 128;                 // tile_dim1   = 128 (bits 143:128), tile_dim2 = 0 -> 2D
    g1[5] = 128;                 // tensor_dim0_stride = 128 (bits 207:160)
    g1[6] = 0;
    g1[7] = 0;
    const i32x4 gz4 = {0, 0, 0, 0};
    const i32x8 gz8 = {0, 0, 0, 0, 0, 0, 0, 0};
    __builtin_amdgcn_tensor_load_to_lds(g0, g1, gz4, gz4, gz8, 0);
  }

  const int rowBase = blockIdx.x * ROWS_PER_WG + wave * 16;
  const float* xrow = x + (size_t)(rowBase + lidx) * IN_DIM;

  float b1v[8], b2v[8];
#pragma unroll
  for (int n = 0; n < 8; ++n) {
    b1v[n] = b1[n * 16 + lidx];
    b2v[n] = b2[n * 16 + lidx];
  }

  // ================= phase 1: i1 = x @ W1^T + b1 =================
  v8f i1[8];
#pragma unroll
  for (int n = 0; n < 8; ++n) {
#pragma unroll
    for (int e = 0; e < 8; ++e) i1[n][e] = b1v[n];
  }

  for (int kc = 0; kc < 25; ++kc) {
    const int k0 = kc * 32;
    half_t* buf = &w1buf[kc & 1][0];
    // cooperative stage of W1[0:128][k0:k0+32] -> f16 LDS (zero-padded K tail)
    if (kc < 24) {
#pragma unroll
      for (int j0 = 0; j0 < 4; ++j0) {
        const int j = j0 * 256 + tid;
        const int r = j >> 3;
        const int k = k0 + (j & 7) * 4;
        const float4 v = *reinterpret_cast<const float4*>(W1 + (size_t)r * IN_DIM + k);
        v4h h; h[0] = (half_t)v.x; h[1] = (half_t)v.y; h[2] = (half_t)v.z; h[3] = (half_t)v.w;
        *reinterpret_cast<v4h*>(buf + j * 4) = h;
      }
    } else {
#pragma unroll
      for (int j0 = 0; j0 < 4; ++j0) {
        const int j = j0 * 256 + tid;
        const int r = j >> 3;
        const int k = k0 + (j & 7) * 4;
        float4 v;
        if (k + 3 < IN_DIM) v = *reinterpret_cast<const float4*>(W1 + (size_t)r * IN_DIM + k);
        else { v.x = 0.f; v.y = 0.f; v.z = 0.f; v.w = 0.f; }
        v4h h; h[0] = (half_t)v.x; h[1] = (half_t)v.y; h[2] = (half_t)v.z; h[3] = (half_t)v.w;
        *reinterpret_cast<v4h*>(buf + j * 4) = h;
      }
    }
    __syncthreads();

    __builtin_prefetch(xrow + k0 + 64, 0, 0);   // global_prefetch_b8 next x chunks
    const bool full = (kc < 24);                // K tail: only 16 of 32 valid in x
    const v16h a = a_frag_global(xrow, k0 + lhalf * 8, full);
    // load ALL B fragments first (distinct regs -> back-to-back ds_loads, staggered waits)
    v16h bf[8];
#pragma unroll
    for (int n = 0; n < 8; ++n)
      bf[n] = a_frag_lds(buf + (n * 16 + lidx) * 32, lhalf * 8);
#pragma unroll
    for (int n = 0; n < 8; ++n)
      i1[n] = __builtin_amdgcn_wmma_f32_16x16x32_f16(false, a, false, bf[n], (short)0, i1[n], false, false);
  }

  // ---- W2 TDM complete -> convert fp32 stage to f16 operand buffer ----
  if (wave == 0) __builtin_amdgcn_s_wait_tensorcnt((short)0);
  __syncthreads();
#pragma unroll
  for (int j0 = 0; j0 < 16; ++j0) {
    const int j = j0 * 256 + tid;
    const float4 v = *reinterpret_cast<const float4*>(&shA.w2stage[j * 4]);
    v4h h; h[0] = (half_t)v.x; h[1] = (half_t)v.y; h[2] = (half_t)v.z; h[3] = (half_t)v.w;
    *reinterpret_cast<v4h*>(w2lds + j * 4) = h;
  }
  __syncthreads();

  // ================= phase 2: LIF time loop =================
  half_t* myS1 = shA.s1 + wave * (16 * H_DIM);
  const half_t* w2b = w2lds + lidx * H_DIM + lhalf * 16;   // + n*16*H_DIM + kc*32
  v8f v1[8], v2[8], spk[8];
#pragma unroll
  for (int n = 0; n < 8; ++n) {
#pragma unroll
    for (int e = 0; e < 8; ++e) { v1[n][e] = 0.0f; v2[n][e] = 0.0f; spk[n][e] = 0.0f; }
  }

  for (int t = 0; t < T_STEPS; ++t) {
    // v1 <- 0.9 v1 + i1 ; s1 = (v1 > 1) ; v1 -= s1 ; stash s1 (f16) in wave-private LDS
#pragma unroll
    for (int n = 0; n < 8; ++n) {
#pragma unroll
      for (int e = 0; e < 8; ++e) {
        const float v = 0.9f * v1[n][e] + i1[n][e];
        const float s = (v > 1.0f) ? 1.0f : 0.0f;
        v1[n][e] = v - s;
        myS1[(e + lhalf * 8) * H_DIM + n * 16 + lidx] = (half_t)s;
      }
    }
    // s1 A-fragments for this wave's 16 rows (LDS in-order per wave: no barrier needed)
    v16h af[4];
#pragma unroll
    for (int kc = 0; kc < 4; ++kc)
      af[kc] = b_frag_lds16(myS1 + lidx * H_DIM + kc * 32 + lhalf * 16);

    // v2 <- 0.9 v2 + b2 + s1 @ W2^T, 2-stage pipelined over n-tiles:
    // W2 fragments for tile n+1 load while tile n's WMMAs execute.
    v16h bf[4];
#pragma unroll
    for (int kc = 0; kc < 4; ++kc)
      bf[kc] = b_frag_lds16(w2b + kc * 32);
#pragma unroll
    for (int n = 0; n < 8; ++n) {
      v16h bfn[4];
      if (n < 7) {
#pragma unroll
        for (int kc = 0; kc < 4; ++kc)
          bfn[kc] = b_frag_lds16(w2b + (n + 1) * 16 * H_DIM + kc * 32);
      }
      v8f acc;
#pragma unroll
      for (int e = 0; e < 8; ++e) acc[e] = 0.9f * v2[n][e] + b2v[n];
#pragma unroll
      for (int kc = 0; kc < 4; ++kc)
        acc = __builtin_amdgcn_wmma_f32_16x16x32_f16(false, af[kc], false, bf[kc], (short)0, acc, false, false);
#pragma unroll
      for (int e = 0; e < 8; ++e) {
        const float s = (acc[e] > 1.0f) ? 1.0f : 0.0f;
        v2[n][e] = acc[e] - s;
        spk[n][e] += s;
      }
      if (n < 7) {
#pragma unroll
        for (int kc = 0; kc < 4; ++kc) bf[kc] = bfn[kc];
      }
    }
  }

  // ================= phase 3: out = (spk/T) @ Wr^T + br =================
#pragma unroll
  for (int n = 0; n < 8; ++n) {
#pragma unroll
    for (int e = 0; e < 8; ++e)
      myS1[(e + lhalf * 8) * H_DIM + n * 16 + lidx] = (half_t)spk[n][e];   // counts <=20: f16-exact
  }
  for (int idx = lane; idx < 16 * OUT_DIM; idx += 32) {
    const int r = idx / OUT_DIM;
    const int o = idx - r * OUT_DIM;
    const half_t* srow = myS1 + r * H_DIM;
    const float*  wrow = Wr + o * H_DIM;
    float acc = 0.0f;
#pragma unroll 4
    for (int h = 0; h < H_DIM; ++h) acc += (float)srow[h] * wrow[h];
    out[(size_t)(rowBase + r) * OUT_DIM + o] = acc * (1.0f / T_STEPS) + br[o];
  }
}

extern "C" void kernel_launch(void* const* d_in, const int* in_sizes, int n_in,
                              void* d_out, int out_size, void* d_ws, size_t ws_size,
                              hipStream_t stream) {
  const float* x  = (const float*)d_in[0];
  const float* W1 = (const float*)d_in[1];
  const float* b1 = (const float*)d_in[2];
  const float* W2 = (const float*)d_in[3];
  const float* b2 = (const float*)d_in[4];
  const float* Wr = (const float*)d_in[5];
  const float* br = (const float*)d_in[6];
  float* out = (float*)d_out;

  const int Brows = in_sizes[0] / IN_DIM;       // 65536
  const int grid  = Brows / ROWS_PER_WG;        // 512
  snn_fused_kernel<<<grid, 256, 0, stream>>>(x, W1, b1, W2, b2, Wr, br, out);
}
